// FlaxDonutSwinSelfAttention_6390911336991
// MI455X (gfx1250) — matrix operand order repaired and checked by reading, not compiled, not run on hardware
//
#include <hip/hip_runtime.h>

#define S   49
#define SP  64
#define DIM 384
#define NH  12
#define HD  32
#define LDH 392   // u16 stride for X/Q/K/V in LDS (384 + 8 pad)
#define LDP 72    // u16 stride for per-head prob buffer (64 + 8 pad)

#define COMB_BYTES (16 * NH * S * S * 4)        // 1,843,968
#define SP_OFF 200704u                           // byte offset of sP region

typedef __attribute__((ext_vector_type(16))) __bf16 v16bf;
typedef __attribute__((ext_vector_type(8)))  float  v8f;

union Frag {
    v16bf v;
    unsigned int d[8];
    unsigned short u[16];
    uint4 q[2];
};

// native bf16 convert (v_cvt on gfx1250)
__device__ inline unsigned short f2bf(float f) {
    union { __bf16 h; unsigned short u; } c;
    c.h = (__bf16)f;
    return c.u;
}
__device__ inline unsigned int pk2(float a, float b) {
    return (unsigned int)f2bf(a) | ((unsigned int)f2bf(b) << 16);
}

// ---------------------------------------------------------------------------
// Prep 1: comb[m][h][q][k] = rel_bias_table[rel_index[q][k]][h] + mask[m][q][k]
// ---------------------------------------------------------------------------
__global__ void prep_comb(const float* __restrict__ mask,
                          const float* __restrict__ table,
                          const int*   __restrict__ relidx,
                          float*       __restrict__ comb) {
    const int m = blockIdx.x / NH;
    const int h = blockIdx.x % NH;
    for (int i = threadIdx.x; i < S * S; i += blockDim.x) {
        const int idx = relidx[i];
        comb[(size_t)(m * NH + h) * (S * S) + i] =
            table[idx * NH + h] + mask[(size_t)m * (S * S) + i];
    }
}

// ---------------------------------------------------------------------------
// Prep 2: convert Wq|Wk|Wv (f32) once into bf16, pre-swizzled into the exact
// 32x16 WMMA B-fragment lane layout:
//   wsw[((ks*72 + ncat/16)*32 + lane)*8 + v] = pack(W[k][n], W[k+1][n])
//   n = (ncat%16 part) , k = ks*32 + (v>>2)*16 + (lane>>4)*8 + 2(v&3)
// Each wave in the main kernel then loads a fragment as 2 coalesced b128s.
// ---------------------------------------------------------------------------
__global__ void prep_w(const float* __restrict__ Wq,
                       const float* __restrict__ Wk,
                       const float* __restrict__ Wv,
                       unsigned int* __restrict__ wsw) {
    const int ks   = blockIdx.x / 72;       // K-step (0..11)
    const int nt   = blockIdx.x % 72;       // 16-col tile over 1152 concat cols
    const int lane = threadIdx.x >> 3;      // 0..31
    const int v    = threadIdx.x & 7;       // 0..7
    const int ncat = nt * 16 + (lane & 15);
    const int mat  = ncat / DIM;
    const int n    = ncat % DIM;
    const float* W = (mat == 0) ? Wq : ((mat == 1) ? Wk : Wv);
    const int k    = ks * 32 + (v >> 2) * 16 + (lane >> 4) * 8 + 2 * (v & 3);
    wsw[(size_t)blockIdx.x * 256 + threadIdx.x] =
        pk2(W[(size_t)k * DIM + n], W[(size_t)(k + 1) * DIM + n]);
}

// ---------------------------------------------------------------------------
// Fused window attention: one block per window (B=1024), 12 waves.
// ---------------------------------------------------------------------------
__global__ __launch_bounds__(384) void swin_attn(
    const float* __restrict__ X,
    const unsigned int* __restrict__ wsw,
    const float* __restrict__ bq,
    const float* __restrict__ bk,
    const float* __restrict__ bv,
    const float* __restrict__ comb,
    float* __restrict__ out) {

    extern __shared__ unsigned char smem[];
    unsigned short* sX = (unsigned short*)(smem);            //  50176 B
    unsigned short* sQ = (unsigned short*)(smem + 50176);    //  50176 B
    unsigned short* sK = (unsigned short*)(smem + 100352);   //  50176 B
    unsigned short* sV = (unsigned short*)(smem + 150528);   //  50176 B
    unsigned short* sP = (unsigned short*)(smem + SP_OFF);   // 110592 B

    const int b     = blockIdx.x;
    const int tid   = threadIdx.x;
    const int lane  = tid & 31;
    const int wave  = tid >> 5;
    const int n16   = lane & 15;
    const int nhalf = lane >> 4;

    // ---------------- Phase A: async-stage X (f32) into LDS, cvt to bf16 ----
    const float* Xb = X + (size_t)b * S * DIM;
    for (int i = tid; i < (S * DIM) / 4; i += 384) {
        const unsigned int ldsa = SP_OFF + (unsigned int)i * 16u;
        const unsigned long long ga =
            (unsigned long long)(const void*)(Xb + (size_t)i * 4);
        asm volatile("global_load_async_to_lds_b128 %0, %1, off"
                     :: "v"(ldsa), "v"(ga) : "memory");
    }
    asm volatile("s_wait_asynccnt 0" ::: "memory");
    __syncthreads();

    const float* fx = (const float*)(smem + SP_OFF);
    for (int i = tid; i < SP * (DIM / 4); i += 384) {
        const int r  = i / (DIM / 4);
        const int c4 = (i % (DIM / 4)) * 4;
        float4 f = make_float4(0.f, 0.f, 0.f, 0.f);
        if (r < S) f = *(const float4*)(fx + r * DIM + c4);
        uint2 p;
        p.x = pk2(f.x, f.y);
        p.y = pk2(f.z, f.w);
        *(uint2*)(sX + r * LDH + c4) = p;
    }
    __syncthreads();

    // ---------------- Phase B: QKV projection -------------------------------
    {
        const int mat = wave >> 2;                       // 0=Q,1=K,2=V
        const float* bias = (mat == 0) ? bq : ((mat == 1) ? bk : bv);
        unsigned short* dst = (mat == 0) ? sQ : ((mat == 1) ? sK : sV);
        const float scl = (mat == 0) ? 0.17677669529663689f : 1.0f;
        const int ncol0 = (wave & 3) * 96;

        #pragma unroll 1
        for (int pass = 0; pass < 2; ++pass) {
            const int n0  = ncol0 + pass * 48;
            const int ntg = (mat * DIM + n0) >> 4;       // global 16-col tile id
            v8f acc[3][4];
            const v8f vzero = {0.f, 0.f, 0.f, 0.f, 0.f, 0.f, 0.f, 0.f};
            #pragma unroll
            for (int nt = 0; nt < 3; ++nt)
                #pragma unroll
                for (int mt = 0; mt < 4; ++mt) acc[nt][mt] = vzero;

            #pragma unroll 1
            for (int ks = 0; ks < 12; ++ks) {
                const int k0 = ks * 32;
                // A fragments (16x32 bf16) from X in LDS
                Frag a[4];
                #pragma unroll
                for (int mt = 0; mt < 4; ++mt) {
                    const unsigned short* p =
                        sX + (mt * 16 + n16) * LDH + k0 + nhalf * 8;
                    a[mt].q[0] = *(const uint4*)(p);
                    a[mt].q[1] = *(const uint4*)(p + 16);
                }
                // B fragments: pre-swizzled bf16 weights, 2 x b128 each
                Frag wb[3];
                #pragma unroll
                for (int nt = 0; nt < 3; ++nt) {
                    const uint4* wp = (const uint4*)(
                        wsw + ((size_t)(ks * 72 + ntg + nt) * 32 + lane) * 8);
                    wb[nt].q[0] = wp[0];
                    wb[nt].q[1] = wp[1];
                }
                if (ks + 1 < 12)
                    __builtin_prefetch(
                        wsw + ((size_t)((ks + 1) * 72 + ntg) * 32 + lane) * 8,
                        0, 1);

                #pragma unroll
                for (int nt = 0; nt < 3; ++nt)
                    #pragma unroll
                    for (int mt = 0; mt < 4; ++mt)
                        acc[nt][mt] = __builtin_amdgcn_wmma_f32_16x16x32_bf16(
                            false, a[mt].v, false, wb[nt].v,
                            (short)0, acc[nt][mt], false, false);
            }
            // bias + (q-scale) and store bf16 result tile to LDS
            #pragma unroll
            for (int nt = 0; nt < 3; ++nt) {
                const int n = n0 + nt * 16 + n16;
                const float bn = bias[n];
                #pragma unroll
                for (int mt = 0; mt < 4; ++mt)
                    #pragma unroll
                    for (int v = 0; v < 8; ++v) {
                        const int r = mt * 16 + v + 8 * nhalf;
                        dst[r * LDH + n] = f2bf((acc[nt][mt][v] + bn) * scl);
                    }
            }
        }
    }
    __syncthreads();

    // ---------------- Phase C: per-head attention ---------------------------
    const int h  = wave;
    const int hc = h * HD;

    Frag aq[4];
    #pragma unroll
    for (int mt = 0; mt < 4; ++mt) {
        const unsigned short* p = sQ + (mt * 16 + n16) * LDH + hc + nhalf * 8;
        aq[mt].q[0] = *(const uint4*)(p);
        aq[mt].q[1] = *(const uint4*)(p + 16);
    }
    Frag bkf[4];   // K^T as B: column n of K^T == row n of K (contiguous)
    #pragma unroll
    for (int nt = 0; nt < 4; ++nt) {
        const unsigned short* p = sK + (nt * 16 + n16) * LDH + hc + nhalf * 8;
        bkf[nt].q[0] = *(const uint4*)(p);
        bkf[nt].q[1] = *(const uint4*)(p + 16);
    }
    v8f s[4][4];
    {
        const v8f vzero = {0.f, 0.f, 0.f, 0.f, 0.f, 0.f, 0.f, 0.f};
        #pragma unroll
        for (int nt = 0; nt < 4; ++nt)
            #pragma unroll
            for (int mt = 0; mt < 4; ++mt)
                s[nt][mt] = __builtin_amdgcn_wmma_f32_16x16x32_bf16(
                    false, aq[mt].v, false, bkf[nt].v,
                    (short)0, vzero, false, false);
    }

    // + (rel_bias + window mask), kill padded key columns
    const float* cb = comb + (size_t)((b & 15) * NH + h) * (S * S);
    #pragma unroll
    for (int nt = 0; nt < 4; ++nt) {
        const int kc = nt * 16 + n16;
        #pragma unroll
        for (int mt = 0; mt < 4; ++mt)
            #pragma unroll
            for (int v = 0; v < 8; ++v) {
                const int qr = mt * 16 + v + 8 * nhalf;
                float x = s[nt][mt][v];
                if (kc < S) { if (qr < S) x += cb[qr * S + kc]; }
                else        { x = -1e30f; }
                s[nt][mt][v] = x;
            }
    }

    // softmax over keys
    unsigned short* ph = sP + h * (SP * LDP);
    #pragma unroll
    for (int mt = 0; mt < 4; ++mt) {
        float mx[8], rs[8];
        #pragma unroll
        for (int v = 0; v < 8; ++v) {
            float m0 = fmaxf(fmaxf(s[0][mt][v], s[1][mt][v]),
                             fmaxf(s[2][mt][v], s[3][mt][v]));
            #pragma unroll
            for (int off = 8; off >= 1; off >>= 1)
                m0 = fmaxf(m0, __shfl_xor(m0, off, 32));
            mx[v] = m0;
        }
        #pragma unroll
        for (int v = 0; v < 8; ++v) {
            float t = 0.f;
            #pragma unroll
            for (int nt = 0; nt < 4; ++nt) {
                const float e = __expf(s[nt][mt][v] - mx[v]);
                s[nt][mt][v] = e;
                t += e;
            }
            #pragma unroll
            for (int off = 8; off >= 1; off >>= 1)
                t += __shfl_xor(t, off, 32);
            rs[v] = 1.0f / t;
        }
        #pragma unroll
        for (int nt = 0; nt < 4; ++nt) {
            const int kc = nt * 16 + n16;
            #pragma unroll
            for (int v = 0; v < 8; ++v) {
                const int qr = mt * 16 + v + 8 * nhalf;
                ph[qr * LDP + kc] = f2bf(s[nt][mt][v] * rs[v]);
            }
        }
    }

    // P as A fragments (2 K-steps of 32), V as B fragments (column gather)
    Frag ap[2][4];
    #pragma unroll
    for (int kt = 0; kt < 2; ++kt)
        #pragma unroll
        for (int mt = 0; mt < 4; ++mt) {
            const unsigned short* p =
                ph + (mt * 16 + n16) * LDP + kt * 32 + nhalf * 8;
            ap[kt][mt].q[0] = *(const uint4*)(p);
            ap[kt][mt].q[1] = *(const uint4*)(p + 16);
        }
    Frag bvf[2][2];
    #pragma unroll
    for (int kt = 0; kt < 2; ++kt)
        #pragma unroll
        for (int nt = 0; nt < 2; ++nt) {
            const int n  = hc + nt * 16 + n16;
            const int kb = kt * 32 + nhalf * 8;
            #pragma unroll
            for (int v = 0; v < 8; ++v) {
                const unsigned int lo = sV[(kb + (v >> 2) * 16 + 2 * (v & 3)) * LDH + n];
                const unsigned int hi = sV[(kb + (v >> 2) * 16 + 2 * (v & 3) + 1) * LDH + n];
                bvf[kt][nt].d[v] = lo | (hi << 16);
            }
        }

    const v8f vzero = {0.f, 0.f, 0.f, 0.f, 0.f, 0.f, 0.f, 0.f};
    #pragma unroll
    for (int mt = 0; mt < 4; ++mt)
        #pragma unroll
        for (int nt = 0; nt < 2; ++nt) {
            v8f o = __builtin_amdgcn_wmma_f32_16x16x32_bf16(
                false, ap[0][mt].v, false, bvf[0][nt].v,
                (short)0, vzero, false, false);
            o = __builtin_amdgcn_wmma_f32_16x16x32_bf16(
                false, ap[1][mt].v, false, bvf[1][nt].v,
                (short)0, o, false, false);
            #pragma unroll
            for (int v = 0; v < 8; ++v) {
                const int qr = mt * 16 + v + 8 * nhalf;
                if (qr < S)
                    out[(size_t)b * S * DIM + (size_t)qr * DIM + hc + nt * 16 + n16] =
                        o[v];
            }
        }
}

extern "C" void kernel_launch(void* const* d_in, const int* in_sizes, int n_in,
                              void* d_out, int out_size, void* d_ws, size_t ws_size,
                              hipStream_t stream) {
    const float* X      = (const float*)d_in[0];
    const float* mask   = (const float*)d_in[1];
    const float* Wq     = (const float*)d_in[2];
    const float* bq     = (const float*)d_in[3];
    const float* Wk     = (const float*)d_in[4];
    const float* bk     = (const float*)d_in[5];
    const float* Wv     = (const float*)d_in[6];
    const float* bv     = (const float*)d_in[7];
    const float* table  = (const float*)d_in[8];
    const int*   relidx = (const int*)d_in[9];
    float*        comb = (float*)d_ws;                          // 1.84 MB
    unsigned int* wsw  = (unsigned int*)((char*)d_ws + COMB_BYTES); // 884 KB
    float* out = (float*)d_out;

    prep_comb<<<16 * NH, 256, 0, stream>>>(mask, table, relidx, comb);
    prep_w<<<12 * 72, 256, 0, stream>>>(Wq, Wk, Wv, wsw);

    const size_t smem = 311296;   // 304 KB of the 320 KB WGP LDS
    (void)hipFuncSetAttribute((const void*)swin_attn,
                              hipFuncAttributeMaxDynamicSharedMemorySize,
                              (int)smem);
    swin_attn<<<1024, 384, smem, stream>>>(X, wsw, bq, bk, bv, comb, out);
}
